// Model_41893111005585
// MI455X (gfx1250) — matrix operand order repaired and checked by reading, not compiled, not run on hardware
//
#include <hip/hip_runtime.h>

namespace {

constexpr int   kNRays   = 1 << 20;
constexpr int   kBlock   = 256;       // 8 waves of 32 (wave32), 256 rays per block
constexpr int   ACT_S    = 72;        // halves per activation row (64 + pad)
constexpr int   W0_S     = 40;        // halves per wT0 row (32 used + pad)
constexpr int   W1_S     = 72;        // halves per wT1/wT2 row (64 used + pad)
constexpr int   W3_S     = 72;        // halves per wT3 row (64 used + pad)
constexpr float kEps     = 1e-9f;
constexpr float kPi      = 3.14159265358979323846f;

typedef _Float16 v8h  __attribute__((ext_vector_type(8)));
typedef _Float16 v16h __attribute__((ext_vector_type(16)));
typedef float    v8f  __attribute__((ext_vector_type(8)));

__device__ __constant__ int      c_res[8] = {4, 8, 19, 43, 95, 210, 463, 1024};
__device__ __constant__ unsigned c_hs[8]  = {256u, 4096u, 130321u, 1048576u,
                                             1048576u, 1048576u, 1048576u, 1048576u};

// Two aligned 16B LDS loads fused into one 16-half WMMA fragment.
__device__ __forceinline__ v16h ld_frag(const _Float16* p, int gap_halves) {
  v8h lo = *(const v8h*)p;
  v8h hi = *(const v8h*)(p + gap_halves);
  return __builtin_shufflevector(lo, hi, 0,1,2,3,4,5,6,7,8,9,10,11,12,13,14,15);
}

// One hidden MLP layer on a 16-ray tile, in place in s_act.
// KB = number of K=32 blocks of the input, NT = number of 16-wide output tiles.
template <int KB, int NT, int WS>
__device__ __forceinline__ void mlp_layer(_Float16* s_act, int rowBase,
                                          const _Float16* wt, const float* bias,
                                          int lane) {
  const int m    = lane & 15;
  const int kSel = lane >> 4;
  const int row  = rowBase + m;

  // Load ALL A fragments before any write-back -> in-place update is wave-safe.
  v16h a[KB];
#pragma unroll
  for (int kb = 0; kb < KB; ++kb)
    a[kb] = ld_frag(&s_act[row * ACT_S + kb * 32 + kSel * 8], 16);

  v8f d[NT];
#pragma unroll
  for (int nt = 0; nt < NT; ++nt) {
    const int col = nt * 16 + m;
    v8f acc = {};
#pragma unroll
    for (int kb = 0; kb < KB; ++kb) {
      v16h b = ld_frag(&wt[col * WS + kb * 32 + kSel * 8], 16);
      acc = __builtin_amdgcn_wmma_f32_16x16x32_f16(false, a[kb], false, b,
                                                   (short)0, acc, false, false);
    }
    const float bv = bias[nt * 16 + m];
#pragma unroll
    for (int v = 0; v < 8; ++v) {
      float r = acc[v] + bv;
      acc[v]  = r > 0.f ? r : 0.f;   // ReLU
    }
    d[nt] = acc;
  }

  // Write back f16 activations (D layout: VGPR v -> row v (+8 for hi lanes)).
#pragma unroll
  for (int nt = 0; nt < NT; ++nt) {
    const int col = nt * 16 + m;
#pragma unroll
    for (int v = 0; v < 8; ++v) {
      const int r = rowBase + v + kSel * 8;
      s_act[r * ACT_S + col] = (_Float16)d[nt][v];
    }
  }
}

// Final layer 64 -> 4 (padded to 16 cols), f32 result straight to global.
__device__ __forceinline__ void mlp_out(const _Float16* s_act, int rowBase,
                                        const _Float16* wt, const float* bias,
                                        int lane, float* __restrict__ out,
                                        int gBase) {
  const int m    = lane & 15;
  const int kSel = lane >> 4;
  const int row  = rowBase + m;
  v8f acc = {};
#pragma unroll
  for (int kb = 0; kb < 2; ++kb) {
    v16h a = ld_frag(&s_act[row * ACT_S + kb * 32 + kSel * 8], 16);
    v16h b = ld_frag(&wt[m * W3_S + kb * 32 + kSel * 8], 16);
    acc = __builtin_amdgcn_wmma_f32_16x16x32_f16(false, a, false, b,
                                                 (short)0, acc, false, false);
  }
  if (m < 4) {                          // divergence only around stores, not WMMA
    const float bv = bias[192 + m];
#pragma unroll
    for (int v = 0; v < 8; ++v) {
      const int r = rowBase + v + kSel * 8;
      out[(size_t)(gBase + r) * 4 + m] = acc[v] + bv;
    }
  }
}

__global__ __launch_bounds__(kBlock)
void plucker_hashgrid_mlp(const float* __restrict__ orig,
                          const float* __restrict__ dirs,
                          const float* __restrict__ t0, const float* __restrict__ t1,
                          const float* __restrict__ t2, const float* __restrict__ t3,
                          const float* __restrict__ t4, const float* __restrict__ t5,
                          const float* __restrict__ t6, const float* __restrict__ t7,
                          const float* __restrict__ w0, const float* __restrict__ b0,
                          const float* __restrict__ w1, const float* __restrict__ b1,
                          const float* __restrict__ w2, const float* __restrict__ b2,
                          const float* __restrict__ w3, const float* __restrict__ b3,
                          float* __restrict__ out) {
  __shared__ _Float16 s_act[kBlock * ACT_S];   // 36.0 KB
  __shared__ _Float16 s_w0[64 * W0_S];         //  5.0 KB (16->64, K padded to 32)
  __shared__ _Float16 s_w1[64 * W1_S];         //  9.0 KB
  __shared__ _Float16 s_w2[64 * W1_S];         //  9.0 KB
  __shared__ _Float16 s_w3[16 * W3_S];         //  2.25 KB (4 cols padded to 16)
  __shared__ float    s_bias[208];

  const int tid   = threadIdx.x;
  const int lane  = tid & 31;
  const int wave  = tid >> 5;
  const int gBase = blockIdx.x * kBlock;

  // ---- Stage 0: zero padded weight regions --------------------------------
  for (int i = tid; i < 64 * W0_S; i += kBlock) s_w0[i] = (_Float16)0.f;
  for (int i = tid; i < 16 * W3_S; i += kBlock) s_w3[i] = (_Float16)0.f;
  __syncthreads();

  // ---- Stage 1: transpose-convert weights f32 -> f16 into LDS -------------
  for (int i = tid; i < 64 * 16; i += kBlock) {       // wT0[n][k] = w0[k,n]
    int n = i >> 4, k = i & 15;
    s_w0[n * W0_S + k] = (_Float16)w0[k * 64 + n];
  }
  for (int i = tid; i < 64 * 64; i += kBlock) {       // wT1/wT2
    int n = i >> 6, k = i & 63;
    s_w1[n * W1_S + k] = (_Float16)w1[k * 64 + n];
    s_w2[n * W1_S + k] = (_Float16)w2[k * 64 + n];
  }
  for (int i = tid; i < 4 * 64; i += kBlock) {        // wT3 (4 valid cols)
    int n = i >> 6, k = i & 63;
    s_w3[n * W3_S + k] = (_Float16)w3[k * 4 + n];
  }
  if (tid < 64) {
    s_bias[tid]       = b0[tid];
    s_bias[64 + tid]  = b1[tid];
    s_bias[128 + tid] = b2[tid];
  }
  if (tid < 16) s_bias[192 + tid] = (tid < 4) ? b3[tid] : 0.f;

  // ---- Stage 2: per-ray Plücker coords + multires hash encoding -----------
  {
    const int ray = gBase + tid;
    const float ox = orig[3 * ray + 0], oy = orig[3 * ray + 1], oz = orig[3 * ray + 2];
    float dx = dirs[3 * ray + 0], dy = dirs[3 * ray + 1], dz = dirs[3 * ray + 2];
    const float dn  = sqrtf(dx * dx + dy * dy + dz * dz);
    const float di  = 1.f / fmaxf(dn, kEps);
    dx *= di; dy *= di; dz *= di;

    const float theta = atan2f(dy, dx);
    const float rxy   = sqrtf(dx * dx + dy * dy);
    const float phi   = atan2f(dz, fmaxf(rxy, kEps));
    const float t01   = (theta + kPi) * (0.5f / kPi);
    const float p01   = (phi + 0.5f * kPi) * (1.f / kPi);

    // c1 = d x e1 = (0, dz, -dy);  c2 = d x e2 = (-dz, 0, dx)
    const float n1 = dz * dz + dy * dy;
    const float n2 = dz * dz + dx * dx;
    const bool  use1 = n1 >= n2;
    float ux = use1 ? 0.f : -dz;
    float uy = use1 ? dz  : 0.f;
    float uz = use1 ? -dy : dx;
    const float ui = 1.f / fmaxf(sqrtf(ux * ux + uy * uy + uz * uz), kEps);
    ux *= ui; uy *= ui; uz *= ui;
    const float vx = dy * uz - dz * uy;
    const float vy = dz * ux - dx * uz;
    const float vz = dx * uy - dy * ux;
    const float mx = oy * dz - oz * dy;
    const float my = oz * dx - ox * dz;
    const float mz = ox * dy - oy * dx;
    const float invR = 1.f / fmaxf(sqrtf(ox * ox + oy * oy + oz * oz), kEps);
    const float mu01 = 0.5f * ((mx * ux + my * uy + mz * uz) * invR + 1.f);
    const float mv01 = 0.5f * ((mx * vx + my * vy + mz * vz) * invR + 1.f);

    const float x01[4] = {t01, p01, mu01, mv01};
    const unsigned PR[4] = {1u, 2654435761u, 805459861u, 3674653429u};
    const float* tabs[8] = {t0, t1, t2, t3, t4, t5, t6, t7};

    _Float16* arow = &s_act[tid * ACT_S];
#pragma unroll
    for (int l = 0; l < 8; ++l) {
      const float    rs = (float)c_res[l];
      const unsigned hs = c_hs[l];
      unsigned hf[4], hc[4];
      float    wf[4], wc[4];
#pragma unroll
      for (int dmn = 0; dmn < 4; ++dmn) {
        const float xs  = x01[dmn] * rs;
        const float fl  = floorf(xs);
        const int   xi  = (int)fl;
        const float xfr = xs - fl;
        hf[dmn] = (unsigned)xi * PR[dmn];
        hc[dmn] = (unsigned)(xi + 1) * PR[dmn];
        wf[dmn] = 1.f - xfr;
        wc[dmn] = xfr;
      }
      float a0 = 0.f, a1 = 0.f;
#pragma unroll
      for (int c = 0; c < 16; ++c) {   // bit set -> ceil corner (mask==False)
        const unsigned h = ((c & 1) ? hc[0] : hf[0]) ^ ((c & 2) ? hc[1] : hf[1]) ^
                           ((c & 4) ? hc[2] : hf[2]) ^ ((c & 8) ? hc[3] : hf[3]);
        const float w = ((c & 1) ? wc[0] : wf[0]) * ((c & 2) ? wc[1] : wf[1]) *
                        ((c & 4) ? wc[2] : wf[2]) * ((c & 8) ? wc[3] : wf[3]);
        const unsigned idx = h % hs;                       // constant-folded per level
        const float2   tv  = *(const float2*)(tabs[l] + 2 * (size_t)idx);
        a0 = fmaf(w, tv.x, a0);
        a1 = fmaf(w, tv.y, a1);
      }
      arow[2 * l + 0] = (_Float16)a0;
      arow[2 * l + 1] = (_Float16)a1;
    }
#pragma unroll
    for (int k = 16; k < 32; ++k) arow[k] = (_Float16)0.f;  // zero-pad K 16..31
  }
  __syncthreads();

  // ---- Stage 3: WMMA MLP, each wave owns 2 tiles of 16 rays ---------------
#pragma unroll
  for (int t = 0; t < 2; ++t) {
    const int rowBase = wave * 32 + t * 16;
    mlp_layer<1, 4, W0_S>(s_act, rowBase, s_w0, s_bias +   0, lane);  // 16 -> 64
    mlp_layer<2, 4, W1_S>(s_act, rowBase, s_w1, s_bias +  64, lane);  // 64 -> 64
    mlp_layer<2, 4, W1_S>(s_act, rowBase, s_w2, s_bias + 128, lane);  // 64 -> 64
    mlp_out(s_act, rowBase, s_w3, s_bias, lane, out, gBase);          // 64 -> 4
  }
}

}  // namespace

extern "C" void kernel_launch(void* const* d_in, const int* in_sizes, int n_in,
                              void* d_out, int out_size, void* d_ws, size_t ws_size,
                              hipStream_t stream) {
  (void)in_sizes; (void)n_in; (void)out_size; (void)d_ws; (void)ws_size;
  const float* orig = (const float*)d_in[0];
  const float* dirs = (const float*)d_in[1];
  const float* t[8];
  for (int i = 0; i < 8; ++i) t[i] = (const float*)d_in[2 + i];
  const float* w0 = (const float*)d_in[10];
  const float* b0 = (const float*)d_in[11];
  const float* w1 = (const float*)d_in[12];
  const float* b1 = (const float*)d_in[13];
  const float* w2 = (const float*)d_in[14];
  const float* b2 = (const float*)d_in[15];
  const float* w3 = (const float*)d_in[16];
  const float* b3 = (const float*)d_in[17];
  float* out = (float*)d_out;

  const int grid = kNRays / kBlock;  // 4096 blocks x 256 threads, exact cover
  plucker_hashgrid_mlp<<<grid, kBlock, 0, stream>>>(
      orig, dirs, t[0], t[1], t[2], t[3], t[4], t[5], t[6], t[7],
      w0, b0, w1, b1, w2, b2, w3, b3, out);
}
